// SimpleSequentialMNISTModel_86474871538200
// MI455X (gfx1250) — matrix operand order repaired and checked by reading, not compiled, not run on hardware
//
#include <hip/hip_runtime.h>
#include <hip/hip_bf16.h>

// CDNA5 / gfx1250: wave32, WMMA 16x16x32 f16 -> f32 accumulate.
typedef __attribute__((ext_vector_type(16))) _Float16 v16h;
typedef __attribute__((ext_vector_type(8)))  _Float16 v8h;
typedef __attribute__((ext_vector_type(8)))  float    v8f;

// All weights (converted to f16) + wave-private activation tiles live in LDS.
// Total = 64512 bytes (< 64KB), one wave (16 batch rows) per workgroup.
struct __align__(16) SMem {
  _Float16 W1[64 * 64];    // r1_bw  [2,32,33] -> rows o_flat=k*32+o, cols 0..32 (pad to 64)
  _Float16 W2[32 * 64];    // r1_ow  [32,64]
  _Float16 W3[128 * 96];   // r2_bw  [2,64,96] -> rows o_flat=k*64+o
  _Float16 W4[64 * 128];   // r2_ow  [64,128]
  _Float16 ci1[16 * 64];   // [x | h1(32) | zero pad]
  _Float16 f1 [16 * 64];   // ns1 flattened (branch-major)
  _Float16 ci2[16 * 96];   // [h1(32) | h2(64)]
  _Float16 f2 [16 * 128];  // ns2 flattened
};

__device__ __forceinline__ v8f wmma_f16(v16h a, v16h b, v8f c) {
  // D = A(16x32 f16) * B(32x16 f16) + C(16x16 f32)
  return __builtin_amdgcn_wmma_f32_16x16x32_f16(false, a, false, b, (short)0, c,
                                                false, false);
}

// A-fragment (16x32 f16, row-major source tile in LDS, stride in halves).
// ISA layout: lanes 0-15 hold row m=lane, K = {k0..k0+7, k0+16..k0+23};
//             lanes 16-31 hold row m=lane-16, K = {k0+8..k0+15, k0+24..k0+31}.
__device__ __forceinline__ v16h load_a_frag(const _Float16* base, int stride,
                                            int k0, int lane) {
  const int m = lane & 15, hi = lane >> 4;
  const _Float16* p = base + m * stride + k0 + hi * 8;
  union { v16h v; v8h h[2]; } u;
  u.h[0] = *(const v8h*)(p);        // ds_load_b128
  u.h[1] = *(const v8h*)(p + 16);   // ds_load_b128
  return u.v;
}

// B-fragment (32x16 f16). Weight tile W is row-major [Nrows][K] in LDS, and
// B[k][n] = W[n][k]. Lane holds column n=lane&15 with K matching the A layout:
// lanes 0-15 -> K={k0..k0+7, k0+16..k0+23}, lanes 16-31 -> K={k0+8..k0+15, k0+24..k0+31}.
__device__ __forceinline__ v16h load_b_frag(const _Float16* W, int stride,
                                            int ntile, int k0, int lane) {
  const int n = lane & 15, hi = lane >> 4;
  const _Float16* p = W + (ntile * 16 + n) * stride + k0 + hi * 8;
  union { v16h v; v8h h[2]; } u;
  u.h[0] = *(const v8h*)(p);
  u.h[1] = *(const v8h*)(p + 16);
  return u.v;
}

// Spill a 16x16 f32 D-fragment to an f16 LDS tile (row-major).
// D layout: VGPR r, lanes 0-15 -> row r, col lane; lanes 16-31 -> row r+8.
__device__ __forceinline__ void store_d_f16(_Float16* base, int stride,
                                            int colbase, int lane, v8f c) {
  const int n = lane & 15, hi = lane >> 4;
  _Float16* p = base + (hi * 8) * stride + colbase + n;
#pragma unroll
  for (int r = 0; r < 8; ++r) p[r * stride] = (_Float16)c[r];
}

__global__ void __launch_bounds__(32)
dhs_rnn_kernel(const float* __restrict__ x,
               const float* __restrict__ r1_bw, const float* __restrict__ r1_bb,
               const float* __restrict__ r1_ow, const float* __restrict__ r1_ob,
               const float* __restrict__ r2_bw, const float* __restrict__ r2_bb,
               const float* __restrict__ r2_ow, const float* __restrict__ r2_ob,
               const float* __restrict__ cw,    const float* __restrict__ cb,
               float* __restrict__ out, int T) {
  __shared__ SMem sm;
  const int lane = threadIdx.x;
  const int row0 = blockIdx.x * 16;   // 16 batch rows per wave/workgroup

  // ---- one-time: weights f32 -> f16 into LDS ------------------------------
  for (int i = lane; i < 64 * 64; i += 32) {
    const int r = i >> 6, c = i & 63;
    sm.W1[i] = (c < 33) ? (_Float16)r1_bw[r * 33 + c] : (_Float16)0.0f;
  }
  for (int i = lane; i < 32 * 64;  i += 32) sm.W2[i] = (_Float16)r1_ow[i];
  for (int i = lane; i < 128 * 96; i += 32) sm.W3[i] = (_Float16)r2_bw[i];
  for (int i = lane; i < 64 * 128; i += 32) sm.W4[i] = (_Float16)r2_ow[i];
  for (int i = lane; i < 16 * 64;  i += 32) sm.ci1[i] = (_Float16)0.0f; // h1=0, pad=0
  for (int i = lane; i < 16 * 96;  i += 32) sm.ci2[i] = (_Float16)0.0f; // h1=h2=0

  // ---- per-lane bias values (column N = lane&15, one per 16-col n-tile) ----
  const int nl = lane & 15;
  float b1[4], b2[2], b3[8], b4[4];
#pragma unroll
  for (int nt = 0; nt < 4; ++nt) b1[nt] = r1_bb[nt * 16 + nl];
#pragma unroll
  for (int nt = 0; nt < 2; ++nt) b2[nt] = r1_ob[nt * 16 + nl];
#pragma unroll
  for (int nt = 0; nt < 8; ++nt) b3[nt] = r2_bb[nt * 16 + nl];
#pragma unroll
  for (int nt = 0; nt < 4; ++nt) b4[nt] = r2_ob[nt * 16 + nl];

  // ---- recurrent state lives in VGPRs as f32 accumulator fragments --------
  v8f s1c[4] = {};   // s1: 16 x 64
  v8f s2c[8] = {};   // s2: 16 x 128
  __syncthreads();

  for (int t = 0; t < T; ++t) {
    // x_t (subsample stride 4) into ci1 column 0
    if (lane < 16)
      sm.ci1[lane * 64] = (_Float16)x[(row0 + lane) * 784 + 4 * t];
    __syncthreads();

    // ---- cell1 branch GEMM: bo1 = ci1[16x64] * W1^T, ns1 = a*s1+(1-a)*bo1 --
    {
      const v16h a0 = load_a_frag(sm.ci1, 64, 0, lane);
      const v16h a1 = load_a_frag(sm.ci1, 64, 32, lane);
#pragma unroll
      for (int nt = 0; nt < 4; ++nt) {
        v8f acc = {};
        acc = wmma_f16(a0, load_b_frag(sm.W1, 64, nt, 0, lane), acc);
        acc = wmma_f16(a1, load_b_frag(sm.W1, 64, nt, 32, lane), acc);
        const float alpha = (nt < 2) ? 0.3f : 0.7f;
        v8f ns;
#pragma unroll
        for (int r = 0; r < 8; ++r)
          ns[r] = alpha * s1c[nt][r] + (1.0f - alpha) * (acc[r] + b1[nt]);
        s1c[nt] = ns;
        store_d_f16(sm.f1, 64, nt * 16, lane, ns);
      }
    }
    __syncthreads();

    // ---- cell1 output GEMM: h1 = relu(f1[16x64] * W2^T + ob1) -------------
    {
      const v16h a0 = load_a_frag(sm.f1, 64, 0, lane);
      const v16h a1 = load_a_frag(sm.f1, 64, 32, lane);
#pragma unroll
      for (int nt = 0; nt < 2; ++nt) {
        v8f acc = {};
        acc = wmma_f16(a0, load_b_frag(sm.W2, 64, nt, 0, lane), acc);
        acc = wmma_f16(a1, load_b_frag(sm.W2, 64, nt, 32, lane), acc);
        v8f h;
#pragma unroll
        for (int r = 0; r < 8; ++r) h[r] = fmaxf(acc[r] + b2[nt], 0.0f);
        store_d_f16(sm.ci1, 64, 1 + nt * 16, lane, h);  // feed next step's ci1
        store_d_f16(sm.ci2, 96, nt * 16, lane, h);      // feed cell2 input
      }
    }
    __syncthreads();

    // ---- cell2 branch GEMM: bo2 = ci2[16x96] * W3^T ------------------------
    {
      const v16h a0 = load_a_frag(sm.ci2, 96, 0, lane);
      const v16h a1 = load_a_frag(sm.ci2, 96, 32, lane);
      const v16h a2 = load_a_frag(sm.ci2, 96, 64, lane);
#pragma unroll
      for (int nt = 0; nt < 8; ++nt) {
        v8f acc = {};
        acc = wmma_f16(a0, load_b_frag(sm.W3, 96, nt, 0, lane), acc);
        acc = wmma_f16(a1, load_b_frag(sm.W3, 96, nt, 32, lane), acc);
        acc = wmma_f16(a2, load_b_frag(sm.W3, 96, nt, 64, lane), acc);
        const float alpha = (nt < 4) ? 0.3f : 0.7f;
        v8f ns;
#pragma unroll
        for (int r = 0; r < 8; ++r)
          ns[r] = alpha * s2c[nt][r] + (1.0f - alpha) * (acc[r] + b3[nt]);
        s2c[nt] = ns;
        store_d_f16(sm.f2, 128, nt * 16, lane, ns);
      }
    }
    __syncthreads();

    // ---- cell2 output GEMM: h2 = relu(f2[16x128] * W4^T + ob2) ------------
    {
      const v16h a0 = load_a_frag(sm.f2, 128, 0, lane);
      const v16h a1 = load_a_frag(sm.f2, 128, 32, lane);
      const v16h a2 = load_a_frag(sm.f2, 128, 64, lane);
      const v16h a3 = load_a_frag(sm.f2, 128, 96, lane);
#pragma unroll
      for (int nt = 0; nt < 4; ++nt) {
        v8f acc = {};
        acc = wmma_f16(a0, load_b_frag(sm.W4, 128, nt, 0, lane), acc);
        acc = wmma_f16(a1, load_b_frag(sm.W4, 128, nt, 32, lane), acc);
        acc = wmma_f16(a2, load_b_frag(sm.W4, 128, nt, 64, lane), acc);
        acc = wmma_f16(a3, load_b_frag(sm.W4, 128, nt, 96, lane), acc);
        v8f h;
#pragma unroll
        for (int r = 0; r < 8; ++r) h[r] = fmaxf(acc[r] + b4[nt], 0.0f);
        store_d_f16(sm.ci2, 96, 32 + nt * 16, lane, h);  // h2 half of ci2
      }
    }
    __syncthreads();
  }

  // ---- classifier: out[16x10] = h2[16x64] * cw^T + cb (tiny; VALU) --------
  {
    const int row = lane & 15;
    const int c0  = (lane >> 4) * 5;   // lanes 0-15 -> cols 0..4, 16-31 -> 5..9
#pragma unroll
    for (int j = 0; j < 5; ++j) {
      const int c = c0 + j;
      float acc = cb[c];
      for (int o = 0; o < 64; ++o)
        acc += (float)sm.ci2[row * 96 + 32 + o] * cw[c * 64 + o];
      out[(row0 + row) * 10 + c] = acc;
    }
  }
}

extern "C" void kernel_launch(void* const* d_in, const int* in_sizes, int n_in,
                              void* d_out, int out_size, void* d_ws, size_t ws_size,
                              hipStream_t stream) {
  (void)n_in; (void)out_size; (void)d_ws; (void)ws_size;
  const float* x     = (const float*)d_in[0];
  const float* r1_bw = (const float*)d_in[1];
  const float* r1_bb = (const float*)d_in[2];
  const float* r1_ow = (const float*)d_in[3];
  const float* r1_ob = (const float*)d_in[4];
  const float* r2_bw = (const float*)d_in[5];
  const float* r2_bb = (const float*)d_in[6];
  const float* r2_ow = (const float*)d_in[7];
  const float* r2_ob = (const float*)d_in[8];
  const float* cw    = (const float*)d_in[9];
  const float* cb    = (const float*)d_in[10];

  const int B = in_sizes[0] / 784;   // 8192
  const int T = 784 / 4;             // 196 subsampled steps

  dhs_rnn_kernel<<<B / 16, 32, 0, stream>>>(
      x, r1_bw, r1_bb, r1_ow, r1_ob, r2_bw, r2_bb, r2_ow, r2_ob, cw, cb,
      (float*)d_out, T);
}
